// GraphAttentionLayer_78967268704567
// MI455X (gfx1250) — matrix operand order repaired and checked by reading, not compiled, not run on hardware
//
#include <hip/hip_runtime.h>
#include <hip/hip_bf16.h>

// ---------------------------------------------------------------------------
// Graph attention layer for gfx1250 (MI455X), wave32 + WMMA f16->f32,
// with double-buffered GLOBAL_LOAD_ASYNC_TO_LDS staging of the L2-resident
// operands (kills the 8x per-wave redundant L2 reads).
// B=4, N=4096, D=128.
// ---------------------------------------------------------------------------

#define B_ 4
#define N_ 4096
#define D_ 128

typedef __attribute__((ext_vector_type(16))) _Float16 v16h;
typedef __attribute__((ext_vector_type(8)))  _Float16 v8h;
typedef __attribute__((ext_vector_type(8)))  float    v8f;
typedef __attribute__((ext_vector_type(4)))  int      v4i;

// ---------------------------------------------------------------------------
// helpers
// ---------------------------------------------------------------------------

static __device__ __forceinline__ v8f zero8() {
    v8f z;
#pragma unroll
    for (int i = 0; i < 8; ++i) z[i] = 0.0f;
    return z;
}

static __device__ __forceinline__ v8f wmma_f16(v16h a, v16h b, v8f c) {
    // v_wmma_f32_16x16x32_f16
    return __builtin_amdgcn_wmma_f32_16x16x32_f16(
        /*neg_a=*/false, a, /*neg_b=*/false, b,
        /*c_mod=*/(short)0, c, /*reuse_a=*/false, /*reuse_b=*/false);
}

// A-fragment (16x32, f16) from a row-major f16 tile with leading dim ld.
// ISA 7.12.2: lanes 0-15 hold M=lane, K = 0..7 / 16..23; lanes 16-31 hold
// M=lane-16, K = 8..15 / 24..31.
static __device__ __forceinline__ v16h load_a_frag(const _Float16* p, int ld, int lane) {
    const int half = (lane >> 4) & 1;
    const int m    = lane & 15;
    const _Float16* row = p + (size_t)m * ld;
    v8h lo = *(const v8h*)(row + half * 8);
    v8h hi = *(const v8h*)(row + 16 + half * 8);
    v16h a;
#pragma unroll
    for (int i = 0; i < 8; ++i) { a[i] = lo[i]; a[8 + i] = hi[i]; }
    return a;
}

// Same A-fragment, converting from a row-major f32 tile.
static __device__ __forceinline__ v16h load_a_frag_f32(const float* p, int ld, int lane) {
    const int half = (lane >> 4) & 1;
    const int m    = lane & 15;
    const float* row = p + (size_t)m * ld;
    v8f lo = *(const v8f*)(row + half * 8);
    v8f hi = *(const v8f*)(row + 16 + half * 8);
    v16h a;
#pragma unroll
    for (int i = 0; i < 8; ++i) { a[i] = (_Float16)lo[i]; a[8 + i] = (_Float16)hi[i]; }
    return a;
}

// B-fragment (32x16, f16) for B = R^T, R row-major with leading dim ld:
// B[k][n] = R[n][k].  Lanes 0-15 -> N=lane, K=0..7 / 16..23; lanes 16-31 ->
// N=lane-16, K=8..15 / 24..31.  Two 16-byte loads per lane.
static __device__ __forceinline__ v16h load_bT_frag(const _Float16* p, int ld, int lane) {
    const int half = (lane >> 4) & 1;
    const int n    = lane & 15;
    const _Float16* row = p + (size_t)n * ld;
    v8h lo = *(const v8h*)(row + half * 8);
    v8h hi = *(const v8h*)(row + 16 + half * 8);
    v16h b;
#pragma unroll
    for (int i = 0; i < 8; ++i) { b[i] = lo[i]; b[8 + i] = hi[i]; }
    return b;
}

// Wave-local LDS ordering (wave reuses only its own LDS region).
static __device__ __forceinline__ void lds_fence_wave() {
    __builtin_amdgcn_wave_barrier();
    asm volatile("s_wait_dscnt 0" ::: "memory");
    __builtin_amdgcn_wave_barrier();
}

// ---- CDNA5 async global->LDS copy (16 bytes per lane) ---------------------
// Builtin signature (from hipcc diagnostic): param0 is a pointer to <4 x i32>
// in the global (AS1 / __device__) address space; LDS side is AS3.
typedef __attribute__((address_space(1))) v4i* gv4i_p;
typedef __attribute__((address_space(3))) v4i* lv4i_p;

static __device__ __forceinline__ void async_cp16(const void* g, void* l) {
#if __has_builtin(__builtin_amdgcn_global_load_async_to_lds_b128)
    __builtin_amdgcn_global_load_async_to_lds_b128(
        (gv4i_p)(void*)g, (lv4i_p)l, 0, 0);
#else
    const unsigned lds_addr = (unsigned)(size_t)l;
    asm volatile("global_load_async_to_lds_b128 %0, %1, off"
                 :: "v"(lds_addr), "v"(g) : "memory");
#endif
}

static __device__ __forceinline__ void wait_async0() {
#if __has_builtin(__builtin_amdgcn_s_wait_asynccnt)
    __builtin_amdgcn_s_wait_asynccnt(0);
#else
    asm volatile("s_wait_asynccnt 0x0" ::: "memory");
#endif
}

// ---------------------------------------------------------------------------
// K1: fused prep GEMM.  out cols: [0,128) features (relu, ->featT f16),
// [128,256) q (scaled, ->q16 f16), [256,384) k (->k16 f16).
// One workgroup = 128 rows, 8 waves x 16 rows.
// ---------------------------------------------------------------------------
__global__ __launch_bounds__(256) void k1_prep(
    const float* __restrict__ H, const float* __restrict__ Wqk,
    const float* __restrict__ bqk, const float* __restrict__ Wpsi,
    const float* __restrict__ bpsi,
    _Float16* __restrict__ q16, _Float16* __restrict__ k16,
    _Float16* __restrict__ featT) {

    __shared__ _Float16 Ht[128 * 128];    // 32 KB
    __shared__ _Float16 WcT[384 * 128];   // 96 KB, WcT[c][d] = W(d, c)

    const int t    = threadIdx.x;
    const int wg   = blockIdx.x;
    const int row0 = wg * 128;            // flat row b*N + n
    const int bIdx = row0 >> 12;
    const int n0   = row0 & (N_ - 1);

    for (int i = t; i < 128 * 128; i += 256) {
        const int r = i >> 7, c = i & 127;
        Ht[i] = (_Float16)H[(size_t)(row0 + r) * D_ + c];
    }
    for (int i = t; i < 384 * 128; i += 256) {
        const int d = i / 384;
        const int c = i - d * 384;
        float wv;
        if (c < 128)      wv = Wpsi[d * D_ + c];
        else              wv = Wqk[d * (2 * D_) + (c - 128)];   // q cols then k cols
        WcT[(size_t)c * 128 + d] = (_Float16)wv;
    }
    __syncthreads();

    const int lane = t & 31, w = t >> 5;
    const int half = lane >> 4, ln = lane & 15;
    const int rowLoc = w * 16 + half * 8;
    const float scale = 1.0f / (sqrtf(128.0f) + 1e-5f);

    v16h ha[4];
#pragma unroll
    for (int kt = 0; kt < 4; ++kt)
        ha[kt] = load_a_frag(&Ht[(w * 16) * 128 + kt * 32], 128, lane);

    for (int ct = 0; ct < 24; ++ct) {
        v16h bf[4];
#pragma unroll
        for (int kt = 0; kt < 4; ++kt)
            bf[kt] = load_bT_frag(&WcT[(ct * 16) * 128 + kt * 32], 128, lane);
        v8f acc = zero8();
#pragma unroll
        for (int kt = 0; kt < 4; ++kt)
            acc = wmma_f16(ha[kt], bf[kt], acc);

        const int c = ct * 16 + ln;                  // 0..383
        if (ct < 8) {                                // features -> featT[b][d][n]
            const float bb = bpsi[c];
            v8h pk;
#pragma unroll
            for (int r = 0; r < 8; ++r)
                pk[r] = (_Float16)fmaxf(acc[r] + bb, 0.0f);
            *(v8h*)(featT + ((size_t)bIdx * D_ + c) * N_ + n0 + rowLoc) = pk;
        } else if (ct < 16) {                        // q, pre-scaled
            const int j = c - 128;
            const float bb = bqk[j];
#pragma unroll
            for (int r = 0; r < 8; ++r)
                q16[(size_t)(row0 + rowLoc + r) * D_ + j] =
                    (_Float16)((acc[r] + bb) * scale);
        } else {                                     // k
            const int j = c - 256;
            const float bb = bqk[128 + j];
#pragma unroll
            for (int r = 0; r < 8; ++r)
                k16[(size_t)(row0 + rowLoc + r) * D_ + j] = (_Float16)(acc[r] + bb);
        }
    }
}

// ---------------------------------------------------------------------------
// K2: per-row softmax stats (max, sum of exp) of S = q * k^T.
// Workgroup = 128 rows of one batch.  k blocks (shared by all 8 waves) are
// staged into LDS with double-buffered async copies.
// ---------------------------------------------------------------------------
__global__ __launch_bounds__(256) void k2_rowstats(
    const _Float16* __restrict__ q16, const _Float16* __restrict__ k16,
    float* __restrict__ rowmax, float* __restrict__ rowsum) {

    __shared__ _Float16 kbuf[2][32 * 128];   // 2 x 8 KB

    const int wg   = blockIdx.x;
    const int row0 = wg * 128;
    const int bIdx = row0 >> 12;
    const int t    = threadIdx.x;
    const int lane = t & 31, w = t >> 5;
    const int half = lane >> 4, ln = lane & 15;
    const int rowsW = row0 + w * 16;

    v16h qa[4];
#pragma unroll
    for (int kt = 0; kt < 4; ++kt)
        qa[kt] = load_a_frag(q16 + (size_t)rowsW * D_ + kt * 32, D_, lane);

    float rmax[8], rsum[8];
#pragma unroll
    for (int r = 0; r < 8; ++r) { rmax[r] = -3.0e38f; rsum[r] = 0.0f; }

    const _Float16* kbase = k16 + (size_t)bIdx * N_ * D_;

    auto stageK = [&](int m0, int s) {
        const char* g = (const char*)(kbase + (size_t)m0 * D_);  // 8 KB contiguous
        char* l = (char*)&kbuf[s][0];
        async_cp16(g + t * 16, l + t * 16);
        async_cp16(g + t * 16 + 4096, l + t * 16 + 4096);
    };

    stageK(0, 0);
    wait_async0();
    __syncthreads();

    for (int m0 = 0; m0 < N_; m0 += 32) {
        const int s = (m0 >> 5) & 1;
        if (m0 + 32 < N_) stageK(m0 + 32, s ^ 1);

#pragma unroll
        for (int j = 0; j < 2; ++j) {
            v16h bf[4];
#pragma unroll
            for (int kt = 0; kt < 4; ++kt)
                bf[kt] = load_bT_frag(&kbuf[s][(j * 16) * 128 + kt * 32], 128, lane);
            v8f acc = zero8();
#pragma unroll
            for (int kt = 0; kt < 4; ++kt)
                acc = wmma_f16(qa[kt], bf[kt], acc);
#pragma unroll
            for (int r = 0; r < 8; ++r) {
                const float v  = acc[r];
                const float nm = fmaxf(rmax[r], v);
                rsum[r] = rsum[r] * __expf(rmax[r] - nm) + __expf(v - nm);
                rmax[r] = nm;
            }
        }
        wait_async0();        // our own staging ops for the other buffer
        __syncthreads();      // everyone done reading buffer s
    }

    // merge across the 16 lanes of each half (same rows, different columns)
#pragma unroll
    for (int dlt = 1; dlt < 16; dlt <<= 1) {
#pragma unroll
        for (int r = 0; r < 8; ++r) {
            const float om = __shfl_xor(rmax[r], dlt, 32);
            const float os = __shfl_xor(rsum[r], dlt, 32);
            const float nm = fmaxf(rmax[r], om);
            rsum[r] = rsum[r] * __expf(rmax[r] - nm) + os * __expf(om - nm);
            rmax[r] = nm;
        }
    }
    if (ln == 0) {
#pragma unroll
        for (int r = 0; r < 8; ++r) {
            const int row = rowsW + half * 8 + r;
            rowmax[row] = rmax[r];
            rowsum[row] = rsum[r];
        }
    }
}

// ---------------------------------------------------------------------------
// K3: masked attention + message aggregation.
// messages[b][m][d] = sum_n exp(S[n,m]-max_n)/sum_n * A[b][n][m] * feat[n][d].
// Workgroup owns a 128-wide m block of one batch; wave owns 16 m rows and a
// 16x128 f32 message tile in registers.  The shared q / featT blocks are
// async-staged into LDS (double buffered); P^T tiles are re-laid-out through
// a 1 KB/wave LDS tile.  Only kernel touching the 256 MB mask (HBM stream).
// ---------------------------------------------------------------------------
__global__ __launch_bounds__(256) void k3_attn_msg(
    const _Float16* __restrict__ q16, const _Float16* __restrict__ k16,
    const _Float16* __restrict__ featT, const float* __restrict__ Amask,
    const float* __restrict__ rowmax, const float* __restrict__ rowsum,
    float* __restrict__ msg) {

    __shared__ _Float16 qbuf[2][32 * 128];   // 2 x 8 KB, rows n0..n0+31
    __shared__ _Float16 fbuf[2][128 * 32];   // 2 x 8 KB, featT[d][n-local]
    __shared__ _Float16 pT[8][16 * 32];      // per-wave P^T staging tile

    const int wg   = blockIdx.x;
    const int bIdx = wg >> 5;
    const int m0   = (wg & 31) * 128;
    const int t    = threadIdx.x;
    const int lane = t & 31, w = t >> 5;
    const int half = lane >> 4, ln = lane & 15;
    const size_t bRow = (size_t)bIdx * N_;
    const int mw = m0 + w * 16;

    v16h ka[4];
#pragma unroll
    for (int kt = 0; kt < 4; ++kt)
        ka[kt] = load_a_frag(k16 + (bRow + mw) * D_ + kt * 32, D_, lane);

    v8f macc[8];
#pragma unroll
    for (int dt = 0; dt < 8; ++dt) macc[dt] = zero8();

    const float* Ab = Amask + bRow * N_;          // A[b][n][m]
    _Float16* pTw = &pT[w][0];
    const _Float16* fbase = featT + (size_t)bIdx * D_ * N_;

    auto stageQF = [&](int nblk, int s) {
        // q rows nblk..nblk+31: 8 KB contiguous
        const char* gq = (const char*)(q16 + (bRow + nblk) * D_);
        char* lq = (char*)&qbuf[s][0];
        async_cp16(gq + t * 16, lq + t * 16);
        async_cp16(gq + t * 16 + 4096, lq + t * 16 + 4096);
        // featT: 128 rows x 64 B at stride N_*2
#pragma unroll
        for (int i = 0; i < 2; ++i) {
            const int c   = t + i * 256;          // 0..511
            const int row = c >> 2;
            const int off = (c & 3) * 16;
            const char* gf = (const char*)(fbase + (size_t)row * N_ + nblk);
            async_cp16(gf + off, (char*)&fbuf[s][row * 32] + off);
        }
    };

    stageQF(0, 0);
    wait_async0();
    __syncthreads();

    for (int n0 = 0; n0 < N_; n0 += 32) {
        const int s = (n0 >> 5) & 1;
        if (n0 + 32 < N_) stageQF(n0 + 32, s ^ 1);

#pragma unroll
        for (int j = 0; j < 2; ++j) {
            const int nT = n0 + j * 16;
            // T[m, n] = sum_d k[m,d] * q_s[n,d]   (S^T tile)
            v16h bf[4];
#pragma unroll
            for (int kt = 0; kt < 4; ++kt)
                bf[kt] = load_bT_frag(&qbuf[s][(j * 16) * 128 + kt * 32], 128, lane);
            v8f tacc = zero8();
#pragma unroll
            for (int kt = 0; kt < 4; ++kt)
                tacc = wmma_f16(ka[kt], bf[kt], tacc);

            const int n = nT + ln;                 // this lane's source row
            const float rm  = rowmax[bRow + n];
            const float inv = 1.0f / rowsum[bRow + n];
            if (n0 + 64 < N_)
                __builtin_prefetch(Ab + (size_t)(n + 64) * N_ + mw, 0, 1);
            const v8f mk = *(const v8f*)(Ab + (size_t)n * N_ + mw + half * 8);
#pragma unroll
            for (int r = 0; r < 8; ++r) {
                const float pv = __expf(tacc[r] - rm) * inv * mk[r];
                pTw[(half * 8 + r) * 32 + j * 16 + ln] = (_Float16)pv;
            }
        }
        lds_fence_wave();                          // pT stores -> cross-lane reads
        const v16h pa = load_a_frag(pTw, 32, lane);
#pragma unroll
        for (int dt = 0; dt < 8; ++dt)
            macc[dt] = wmma_f16(pa,
                                load_bT_frag(&fbuf[s][(dt * 16) * 32], 32, lane),
                                macc[dt]);
        wait_async0();        // our staging for the other buffer has landed
        __syncthreads();      // all waves done reading buffer s (and pT WAR)
    }

    float* mrow = msg + (bRow + mw) * D_;
#pragma unroll
    for (int dt = 0; dt < 8; ++dt)
#pragma unroll
        for (int r = 0; r < 8; ++r)
            mrow[(size_t)(half * 8 + r) * D_ + dt * 16 + ln] = macc[dt][r];
}

// ---------------------------------------------------------------------------
// K4: out = relu([messages | H] @ W1 + b1) @ W2 + b2.
// K=256 first GEMM: kt 0..3 from msg (f32 ws), kt 4..7 from H (f32 input).
// ---------------------------------------------------------------------------
__global__ __launch_bounds__(256) void k4_phi(
    const float* __restrict__ msg, const float* __restrict__ H,
    const float* __restrict__ W1, const float* __restrict__ b1,
    const float* __restrict__ W2, const float* __restrict__ b2,
    float* __restrict__ out) {

    __shared__ _Float16 W1T[128 * 256];   // 64 KB, W1T[j][d] = W1[d][j], d<256
    __shared__ _Float16 W2T[128 * 128];   // 32 KB, W2T[j][d] = W2[d][j]
    __shared__ _Float16 hT[128 * 128];    // 32 KB

    const int t    = threadIdx.x;
    const int wg   = blockIdx.x;
    const int row0 = wg * 128;

    for (int i = t; i < 256 * 128; i += 256) {      // i = d*128 + j
        const int j = i & 127, d = i >> 7;
        W1T[(size_t)j * 256 + d] = (_Float16)W1[i];
    }
    for (int i = t; i < 128 * 128; i += 256) {
        const int j = i & 127, d = i >> 7;
        W2T[(size_t)j * 128 + d] = (_Float16)W2[i];
    }
    __syncthreads();

    const int lane = t & 31, w = t >> 5;
    const int half = lane >> 4, ln = lane & 15;

    // A-fragments are ct-invariant: load once.
    v16h a1[8];
#pragma unroll
    for (int kt = 0; kt < 4; ++kt)
        a1[kt] = load_a_frag_f32(msg + (size_t)(row0 + w * 16) * D_ + kt * 32, D_, lane);
#pragma unroll
    for (int kt = 0; kt < 4; ++kt)
        a1[4 + kt] = load_a_frag_f32(H + (size_t)(row0 + w * 16) * D_ + kt * 32, D_, lane);

    // GEMM1: h = relu(msg @ W1[:128] + H @ W1[128:] + b1)
    for (int ct = 0; ct < 8; ++ct) {
        v16h bf[8];
#pragma unroll
        for (int kt = 0; kt < 8; ++kt)
            bf[kt] = load_bT_frag(&W1T[(ct * 16) * 256 + kt * 32], 256, lane);
        v8f acc = zero8();
#pragma unroll
        for (int kt = 0; kt < 8; ++kt)
            acc = wmma_f16(a1[kt], bf[kt], acc);

        const int c = ct * 16 + ln;
        const float bb = b1[c];
#pragma unroll
        for (int r = 0; r < 8; ++r) {
            const float hv = fmaxf(acc[r] + bb, 0.0f);
            hT[(size_t)(w * 16 + half * 8 + r) * 128 + c] = (_Float16)hv;
        }
    }
    lds_fence_wave();   // wave-local rows only

    v16h ah[4];
#pragma unroll
    for (int kt = 0; kt < 4; ++kt)
        ah[kt] = load_a_frag(&hT[(w * 16) * 128 + kt * 32], 128, lane);

    // GEMM2: out = h @ W2 + b2
    for (int ct = 0; ct < 8; ++ct) {
        v16h bf[4];
#pragma unroll
        for (int kt = 0; kt < 4; ++kt)
            bf[kt] = load_bT_frag(&W2T[(ct * 16) * 128 + kt * 32], 128, lane);
        v8f acc = zero8();
#pragma unroll
        for (int kt = 0; kt < 4; ++kt)
            acc = wmma_f16(ah[kt], bf[kt], acc);

        const int c = ct * 16 + ln;
        const float bb = b2[c];
#pragma unroll
        for (int r = 0; r < 8; ++r)
            out[(size_t)(row0 + w * 16 + half * 8 + r) * D_ + c] = acc[r] + bb;
    }
}

// ---------------------------------------------------------------------------
// launch
// ---------------------------------------------------------------------------
extern "C" void kernel_launch(void* const* d_in, const int* in_sizes, int n_in,
                              void* d_out, int out_size, void* d_ws, size_t ws_size,
                              hipStream_t stream) {
    (void)in_sizes; (void)n_in; (void)out_size; (void)ws_size;
    const float* H    = (const float*)d_in[0];
    const float* A    = (const float*)d_in[1];
    const float* Wqk  = (const float*)d_in[2];
    const float* bqk  = (const float*)d_in[3];
    const float* Wpsi = (const float*)d_in[4];
    const float* bpsi = (const float*)d_in[5];
    const float* W1   = (const float*)d_in[6];
    const float* b1   = (const float*)d_in[7];
    const float* W2   = (const float*)d_in[8];
    const float* b2   = (const float*)d_in[9];
    float* out = (float*)d_out;

    char* ws = (char*)d_ws;
    _Float16* q16   = (_Float16*)(ws);                                   // 4 MB
    _Float16* k16   = (_Float16*)(ws + ((size_t)4 << 20));               // 4 MB
    _Float16* featT = (_Float16*)(ws + ((size_t)8 << 20));               // 4 MB
    float*    rowmax = (float*)(ws + ((size_t)12 << 20));                // 64 KB
    float*    rowsum = (float*)(ws + ((size_t)12 << 20) + (64u << 10));  // 64 KB
    float*    msg    = (float*)(ws + ((size_t)13 << 20));                // 8 MB

    const dim3 blk(256);
    const int nTiles = (B_ * N_) / 128;   // 128
    k1_prep<<<nTiles, blk, 0, stream>>>(H, Wqk, bqk, Wpsi, bpsi, q16, k16, featT);
    k2_rowstats<<<nTiles, blk, 0, stream>>>(q16, k16, rowmax, rowsum);
    k3_attn_msg<<<nTiles, blk, 0, stream>>>(q16, k16, featT, A, rowmax, rowsum, msg);
    k4_phi<<<nTiles, blk, 0, stream>>>(msg, H, W1, b1, W2, b2, out);
}
